// LearnableStatPoolingMLP_25417616458336
// MI455X (gfx1250) — matrix operand order repaired and checked by reading, not compiled, not run on hardware
//
#include <hip/hip_runtime.h>
#include <hip/hip_bf16.h>

typedef __attribute__((ext_vector_type(16))) __bf16       v16bf;
typedef __attribute__((ext_vector_type(8)))  float        v8f;
typedef __attribute__((ext_vector_type(4)))  unsigned int v4u;
typedef __attribute__((ext_vector_type(8)))  int          v8i;
typedef __attribute__((ext_vector_type(4)))  int          v4i;

#define B_   32
#define S_   512
#define L_   13
#define F_   768
#define H_   512
#define H2_  256
#define F2_  1536
#define SCHUNKS 16
#define SPERCHUNK (S_/SCHUNKS)

#define KT1 (F2_/32)        // 48 k-tiles, layer 1
#define NT1 (H_/16)         // 32 n-tiles, layer 1
#define KT2 (H_/32)         // 16 k-tiles, layer 2
#define NT2 (H2_/16)        // 16 n-tiles, layer 2
#define SLAB1 (32*H_)       // bf16 elems per layer-1 B k-slab (32 KB)
#define SLAB2 (32*H2_)      // bf16 elems per layer-2 B k-slab (16 KB)

#if defined(__has_builtin)
#if __has_builtin(__builtin_amdgcn_tensor_load_to_lds) && __has_builtin(__builtin_amdgcn_s_wait_tensorcnt)
#define USE_TDM 1
#endif
#endif
#ifndef USE_TDM
#define USE_TDM 0
#endif

// ---------------------------------------------------------------------------
// Phase 1: fused layer-softmax contraction + masked sum / sumsq partials.
// Positions s >= lengths[b] contribute nothing -> skipped entirely.
// ---------------------------------------------------------------------------
__global__ __launch_bounds__(192) void k_reduce(const float* __restrict__ x,
                                                const int*   __restrict__ lengths,
                                                const float* __restrict__ lw,
                                                float* __restrict__ psum,
                                                float* __restrict__ psumsq) {
    const int chunk = blockIdx.x;
    const int b     = blockIdx.y;
    const int t     = threadIdx.x;

    float wl[L_];
    float mx = lw[0];
    #pragma unroll
    for (int l = 1; l < L_; ++l) mx = fmaxf(mx, lw[l]);
    float sw = 0.f;
    #pragma unroll
    for (int l = 0; l < L_; ++l) { wl[l] = __expf(lw[l] - mx); sw += wl[l]; }
    const float invsw = 1.f / sw;
    #pragma unroll
    for (int l = 0; l < L_; ++l) wl[l] *= invsw;

    const int len = lengths[b];
    const int s0  = chunk * SPERCHUNK;
    const int s1  = min(s0 + SPERCHUNK, len);

    float sx = 0.f, sy = 0.f, sz = 0.f, sw4 = 0.f;
    float qx = 0.f, qy = 0.f, qz = 0.f, qw  = 0.f;

    const size_t fo = (size_t)t * 4;
    for (int s = s0; s < s1; ++s) {
        const float* base = x + ((size_t)(b * S_ + s) * L_) * F_ + fo;
        __builtin_prefetch(base + (size_t)L_ * F_, 0, 1);
        float wx = 0.f, wy = 0.f, wz = 0.f, ww = 0.f;
        #pragma unroll
        for (int l = 0; l < L_; ++l) {
            const float4 v = *(const float4*)(base + l * F_);
            wx += v.x * wl[l]; wy += v.y * wl[l];
            wz += v.z * wl[l]; ww += v.w * wl[l];
        }
        sx += wx; sy += wy; sz += wz; sw4 += ww;
        qx += wx * wx; qy += wy * wy; qz += wz * wz; qw += ww * ww;
    }
    const size_t o = (size_t)(b * SCHUNKS + chunk) * F_ + fo;
    float4 s4 = {sx, sy, sz, sw4};
    float4 q4 = {qx, qy, qz, qw};
    *(float4*)(psum   + o) = s4;
    *(float4*)(psumsq + o) = q4;
}

// ---------------------------------------------------------------------------
// A-fragment addressing (16-bit A 16x32, ISA 7.12.2):
//   lanes 0-15 : elems 0..7 -> K 0..7,  elems 8..15 -> K 16..23
//   lanes 16-31: elems 0..7 -> K 8..15, elems 8..15 -> K 24..31
// Fragment buffer order: ((mtile*KT + ktile)*32 + lane)*16 + elem
// ---------------------------------------------------------------------------
__device__ __forceinline__ void store_feat4(__bf16* __restrict__ Af, int b, int K0,
                                            float f0, float f1, float f2, float f3) {
    const int mt = b >> 4, r = b & 15;
    const int kt = K0 >> 5, ko = K0 & 31;
    const int half = (ko >> 3) & 1;
    const int e    = (ko & 7) + ((ko >> 4) << 3);
    __bf16* p = Af + (((size_t)(mt * KT1 + kt) * 32 + r + 16 * half) << 4) + e;
    p[0] = (__bf16)f0; p[1] = (__bf16)f1; p[2] = (__bf16)f2; p[3] = (__bf16)f3;
}

// ---------------------------------------------------------------------------
// Phase 2: combine partials -> mean / unbiased std -> bf16 feats, written
// directly in layer-1 A-fragment order.
// ---------------------------------------------------------------------------
__global__ __launch_bounds__(192) void k_finalize(const float* __restrict__ psum,
                                                  const float* __restrict__ psumsq,
                                                  const int*   __restrict__ lengths,
                                                  __bf16* __restrict__ Afeat) {
    const int b = blockIdx.x;
    const int t = threadIdx.x;
    const size_t fo = (size_t)t * 4;

    float sx = 0.f, sy = 0.f, sz = 0.f, sw = 0.f;
    float qx = 0.f, qy = 0.f, qz = 0.f, qw = 0.f;
    #pragma unroll
    for (int c = 0; c < SCHUNKS; ++c) {
        const size_t o = (size_t)(b * SCHUNKS + c) * F_ + fo;
        const float4 s4 = *(const float4*)(psum   + o);
        const float4 q4 = *(const float4*)(psumsq + o);
        sx += s4.x; sy += s4.y; sz += s4.z; sw += s4.w;
        qx += q4.x; qy += q4.y; qz += q4.z; qw += q4.w;
    }
    const float n     = (float)lengths[b];
    const float invn  = 1.f / fmaxf(n, 1.f);
    const float invn1 = 1.f / fmaxf(n - 1.f, 1.f);

    const float m0 = sx * invn, m1 = sy * invn, m2 = sz * invn, m3 = sw * invn;
    const float v0 = fmaxf((qx - n * m0 * m0) * invn1, 0.f);
    const float v1 = fmaxf((qy - n * m1 * m1) * invn1, 0.f);
    const float v2 = fmaxf((qz - n * m2 * m2) * invn1, 0.f);
    const float v3 = fmaxf((qw - n * m3 * m3) * invn1, 0.f);
    const bool  big = n > 1.f;
    const float d0 = big ? sqrtf(v0) : 0.f;
    const float d1 = big ? sqrtf(v1) : 0.f;
    const float d2 = big ? sqrtf(v2) : 0.f;
    const float d3 = big ? sqrtf(v3) : 0.f;

    const int f0 = t * 4;
    store_feat4(Afeat, b, f0,      m0, m1, m2, m3);   // mean -> K = f
    store_feat4(Afeat, b, f0 + F_, d0, d1, d2, d3);   // std  -> K = f + 768
}

// ---------------------------------------------------------------------------
// Weight conversion fp32 -> bf16, scattered into B-fragment order:
//   lanes 0-15 : col N=lane,    elems 0..7 -> K 0..7,  elems 8..15 -> K 16..23
//   lanes 16-31: col N=lane-16, elems 0..7 -> K 8..15, elems 8..15 -> K 24..31
// Flat order ((ktile*NT + ntile)*32 + lane)*16 + elem  ->  k-slabs contiguous.
// ---------------------------------------------------------------------------
__global__ void k_convert(const float* __restrict__ W1, const float* __restrict__ W2,
                          __bf16* __restrict__ W1b, __bf16* __restrict__ W2b) {
    const int i = blockIdx.x * 256 + threadIdx.x;
    if (i < F2_ * H_) {
        const int e = i & 15, lane = (i >> 4) & 31, tile = i >> 9;
        const int nt = tile & (NT1 - 1), kt = tile / NT1;
        const int K = kt * 32 + ((lane >= 16) ? 8 : 0) + (e & 7) + ((e >> 3) << 4);
        const int N = nt * 16 + (lane & 15);
        W1b[i] = (__bf16)W1[(size_t)K * H_ + N];
    }
    if (i < H_ * H2_) {
        const int e = i & 15, lane = (i >> 4) & 31, tile = i >> 9;
        const int nt = tile & (NT2 - 1), kt = tile / NT2;
        const int K = kt * 32 + ((lane >= 16) ? 8 : 0) + (e & 7) + ((e >> 3) << 4);
        const int N = nt * 16 + (lane & 15);
        W2b[i] = (__bf16)W2[(size_t)K * H2_ + N];
    }
}

// ---------------------------------------------------------------------------
// WMMA helpers
// ---------------------------------------------------------------------------
__device__ __forceinline__ v8f wmma_bf16(v16bf a, v16bf b, v8f c) {
    return __builtin_amdgcn_wmma_f32_16x16x32_bf16(
        false, a, false, b, (short)0, c, false, false);
}

// C/D 16x16 f32: lanes 0-15 -> N=lane, VGPR i -> M=i; lanes 16-31 -> M=8+i.
// Writeback of layer-1 C tiles into *layer-2 A-fragment order* (LDS).
__device__ __forceinline__ void store_c_relu_fragA(__bf16* __restrict__ H1f,
                                                   int m0, int n0, int lane, v8f c,
                                                   const float* __restrict__ bias) {
    const int col   = n0 + (lane & 15);          // this is K for layer 2
    const int rbase = (lane < 16) ? 0 : 8;
    const int mt = m0 >> 4;
    const int kt = col >> 5, ko = col & 31;
    const int half = (ko >> 3) & 1;
    const int e    = (ko & 7) + ((ko >> 4) << 3);
    const float bv = bias[col];
    __bf16* base = H1f + (((size_t)(mt * KT2 + kt) * 32 + 16 * half) << 4) + e;
    #pragma unroll
    for (int i = 0; i < 8; ++i) {
        float v = c[i] + bv;
        base[(size_t)(rbase + i) << 4] = (__bf16)(v > 0.f ? v : 0.f);
    }
}

// Plain row-major C writeback (layer 2 -> H2, consumed row-wise by layer 3)
__device__ __forceinline__ void store_c_relu(__bf16* __restrict__ Hout, int ldh,
                                             int m0, int n0, int lane, v8f c,
                                             const float* __restrict__ bias) {
    const int col   = n0 + (lane & 15);
    const int rbase = m0 + ((lane < 16) ? 0 : 8);
    const float bv  = bias[col];
    #pragma unroll
    for (int i = 0; i < 8; ++i) {
        float v = c[i] + bv;
        Hout[(size_t)(rbase + i) * ldh + col] = (__bf16)(v > 0.f ? v : 0.f);
    }
}

#if USE_TDM
// TDM: 1-D tile copy of `elems` bf16 from global -> LDS (D# per ISA §8.3/8.4).
// This toolchain exposes the 6-arg builtin:
//   (uint32x4 g0, int32x8 g1, int32x4 g2, int32x4 g3, int32x8 pad, i32 cpol)
__device__ __forceinline__ void tdm_load_slab(unsigned lds_byte_off,
                                              const __bf16* gsrc, unsigned elems) {
    const unsigned long long ga = (unsigned long long)(uintptr_t)gsrc;
    v4u g0;
    g0[0] = 1u;                                           // count=1, user mode
    g0[1] = lds_byte_off;                                 // lds_addr
    g0[2] = (unsigned)(ga & 0xFFFFFFFFu);                 // global_addr[31:0]
    g0[3] = (unsigned)((ga >> 32) & 0x01FFFFFFu) | (2u << 30); // addr[56:32], type=2
    v8i g1;
    g1[0] = (int)(1u << 16);                              // data_size=1 (2B)
    g1[1] = (int)((elems & 0xFFFFu) << 16);               // tensor_dim0[15:0]
    g1[2] = (int)((elems >> 16) & 0xFFFFu) | (1 << 16);   // tensor_dim0[31:16], dim1=1
    g1[3] = (int)((elems & 0xFFFFu) << 16);               // tile_dim0
    g1[4] = 0;                                            // tile_dim1/2 unused
    g1[5] = (int)elems;                                   // tensor_dim0_stride
    g1[6] = 0;
    g1[7] = 0;
    const v4i z4 = {0, 0, 0, 0};
    const v8i z8 = {0, 0, 0, 0, 0, 0, 0, 0};
    __builtin_amdgcn_tensor_load_to_lds(g0, g1, z4, z4, z8, 0);
}
#endif

// ---------------------------------------------------------------------------
// Phase 3: MLP. One workgroup, 8 wave32 waves, bf16 WMMA w/ fp32 accumulate.
// B k-slabs double-buffered into LDS via the Tensor Data Mover.
// Dynamic LDS: H1f 32KB | H2 16KB | Bbuf0 32KB | Bbuf1 32KB = 112KB.
// ---------------------------------------------------------------------------
__global__ __launch_bounds__(256) void k_mlp(const __bf16* __restrict__ Afeat,
                                             const __bf16* __restrict__ W1b,
                                             const float*  __restrict__ b1,
                                             const __bf16* __restrict__ W2b,
                                             const float*  __restrict__ b2,
                                             const float*  __restrict__ W3,
                                             const float*  __restrict__ b3,
                                             float* __restrict__ out) {
    extern __shared__ __align__(16) char smem[];
    __bf16* H1f  = (__bf16*)smem;                    // layer-2 A-fragment order
    __bf16* H2s  = (__bf16*)(smem + 32768);          // row-major [32][256]
    __bf16* Bb0  = (__bf16*)(smem + 49152);
    __bf16* Bb1  = (__bf16*)(smem + 49152 + 32768);
    __bf16* Bbuf[2] = {Bb0, Bb1};

    const int tid  = threadIdx.x;
    const int lane = tid & 31;
    const int w    = tid >> 5;
    const v16bf* Av = (const v16bf*)Afeat;

    // ---- layer 1: [32,1536] @ [1536,512], ReLU -> H1f ----
    v8f acc[2][4];
    #pragma unroll
    for (int m = 0; m < 2; ++m)
        #pragma unroll
        for (int j = 0; j < 4; ++j)
            #pragma unroll
            for (int e = 0; e < 8; ++e) acc[m][j][e] = 0.f;

#if USE_TDM
    if (w == 0) tdm_load_slab((unsigned)(uintptr_t)Bbuf[0], W1b, SLAB1);
#endif
    for (int kt = 0; kt < KT1; ++kt) {
#if USE_TDM
        if (w == 0) {
            if (kt + 1 < KT1) {
                tdm_load_slab((unsigned)(uintptr_t)Bbuf[(kt + 1) & 1],
                              W1b + (size_t)(kt + 1) * SLAB1, SLAB1);
                __builtin_amdgcn_s_wait_tensorcnt(1);   // slab kt has landed
            } else {
                __builtin_amdgcn_s_wait_tensorcnt(0);
            }
        }
        __syncthreads();
#else
        {   // cooperative fallback copy of slab kt
            const uint4* src = (const uint4*)(W1b + (size_t)kt * SLAB1);
            uint4* dst = (uint4*)Bbuf[kt & 1];
            for (int i = tid; i < SLAB1 * 2 / 16; i += 256) dst[i] = src[i];
        }
        __syncthreads();
#endif
        const v16bf* Bv = (const v16bf*)Bbuf[kt & 1];
        const v16bf a0 = Av[(size_t)(0 * KT1 + kt) * 32 + lane];
        const v16bf a1 = Av[(size_t)(1 * KT1 + kt) * 32 + lane];
        #pragma unroll
        for (int j = 0; j < 4; ++j) {
            const int nt = w * 4 + j;
            const v16bf bf = Bv[(size_t)nt * 32 + lane];
            acc[0][j] = wmma_bf16(a0, bf, acc[0][j]);
            acc[1][j] = wmma_bf16(a1, bf, acc[1][j]);
        }
        __syncthreads();
    }
    #pragma unroll
    for (int m = 0; m < 2; ++m)
        #pragma unroll
        for (int j = 0; j < 4; ++j)
            store_c_relu_fragA(H1f, m * 16, (w * 4 + j) * 16, lane, acc[m][j], b1);
    __syncthreads();

    // ---- layer 2: [32,512] @ [512,256], ReLU -> H2s ----
    v8f acc2[2][2];
    #pragma unroll
    for (int m = 0; m < 2; ++m)
        #pragma unroll
        for (int j = 0; j < 2; ++j)
            #pragma unroll
            for (int e = 0; e < 8; ++e) acc2[m][j][e] = 0.f;

    const v16bf* H1v = (const v16bf*)H1f;
#if USE_TDM
    if (w == 0) tdm_load_slab((unsigned)(uintptr_t)Bbuf[0], W2b, SLAB2);
#endif
    for (int kt = 0; kt < KT2; ++kt) {
#if USE_TDM
        if (w == 0) {
            if (kt + 1 < KT2) {
                tdm_load_slab((unsigned)(uintptr_t)Bbuf[(kt + 1) & 1],
                              W2b + (size_t)(kt + 1) * SLAB2, SLAB2);
                __builtin_amdgcn_s_wait_tensorcnt(1);
            } else {
                __builtin_amdgcn_s_wait_tensorcnt(0);
            }
        }
        __syncthreads();
#else
        {
            const uint4* src = (const uint4*)(W2b + (size_t)kt * SLAB2);
            uint4* dst = (uint4*)Bbuf[kt & 1];
            for (int i = tid; i < SLAB2 * 2 / 16; i += 256) dst[i] = src[i];
        }
        __syncthreads();
#endif
        const v16bf* Bv = (const v16bf*)Bbuf[kt & 1];
        const v16bf a0 = H1v[(size_t)(0 * KT2 + kt) * 32 + lane];
        const v16bf a1 = H1v[(size_t)(1 * KT2 + kt) * 32 + lane];
        #pragma unroll
        for (int j = 0; j < 2; ++j) {
            const int nt = w * 2 + j;
            const v16bf bf = Bv[(size_t)nt * 32 + lane];
            acc2[0][j] = wmma_bf16(a0, bf, acc2[0][j]);
            acc2[1][j] = wmma_bf16(a1, bf, acc2[1][j]);
        }
        __syncthreads();
    }
    #pragma unroll
    for (int m = 0; m < 2; ++m)
        #pragma unroll
        for (int j = 0; j < 2; ++j)
            store_c_relu(H2s, H2_, m * 16, (w * 2 + j) * 16, lane, acc2[m][j], b2);
    __syncthreads();

    // ---- layer 3: [32,256] @ [256,1] + b3 ----
    if (tid < B_) {
        float a = b3[0];
        const __bf16* hrow = H2s + (size_t)tid * H2_;
        #pragma unroll 8
        for (int k = 0; k < H2_; ++k) a += (float)hrow[k] * W3[k];
        out[tid] = a;
    }
}

// ---------------------------------------------------------------------------
extern "C" void kernel_launch(void* const* d_in, const int* in_sizes, int n_in,
                              void* d_out, int out_size, void* d_ws, size_t ws_size,
                              hipStream_t stream) {
    const float* x       = (const float*)d_in[0];
    const int*   lengths = (const int*)  d_in[1];
    const float* lw      = (const float*)d_in[2];
    const float* W1      = (const float*)d_in[3];
    const float* b1      = (const float*)d_in[4];
    const float* W2      = (const float*)d_in[5];
    const float* b2      = (const float*)d_in[6];
    const float* W3      = (const float*)d_in[7];
    const float* b3      = (const float*)d_in[8];
    float* out = (float*)d_out;

    // workspace layout (~5.1 MB)
    float*  psum   = (float*)d_ws;                                   // 393216 f32
    float*  psumsq = psum + (size_t)B_ * SCHUNKS * F_;               // 393216 f32
    __bf16* Afeat  = (__bf16*)(psumsq + (size_t)B_ * SCHUNKS * F_);  // 49152 bf16
    __bf16* W1b    = Afeat + (size_t)B_ * F2_;                       // 786432 bf16
    __bf16* W2b    = W1b + (size_t)F2_ * H_;                         // 131072 bf16

    k_reduce  <<<dim3(SCHUNKS, B_), 192, 0, stream>>>(x, lengths, lw, psum, psumsq);
    k_convert <<<(F2_ * H_ + 255) / 256, 256, 0, stream>>>(W1, W2, W1b, W2b);
    k_finalize<<<B_, 192, 0, stream>>>(psum, psumsq, lengths, Afeat);
    k_mlp     <<<1, 256, 114688, stream>>>(Afeat, W1b, b1, W2b, b2, W3, b3, out);
}
